// BertWithLayer_8985071583325
// MI455X (gfx1250) — matrix operand order repaired and checked by reading, not compiled, not run on hardware
//
#include <hip/hip_runtime.h>
#include <hip/hip_bf16.h>

// BERT-with-layer scan for MI455X (gfx1250), bf16 WMMA path.
// Dead code eliminated: seq-len-1 softmax == 1.0 exactly, so q/k are unused.
// Weights packed once into WMMA B-fragment layout (contiguous 32B per lane),
// activations staged in LDS row-major bf16 (A fragment = 2x ds_load_b128).
// Pure-copy LDS staging uses GLOBAL_LOAD_ASYNC_TO_LDS_B128 (ASYNCcnt) when the
// toolchain declares the builtin.

#define B_   128
#define S_   512
#define D_   768
#define FF_  3072
#define OUT_ 10
#define EPS_ 1e-12f

typedef __attribute__((ext_vector_type(16))) __bf16 v16bf;
typedef __attribute__((ext_vector_type(8)))  __bf16 v8bf;
typedef __attribute__((ext_vector_type(8)))  float  v8f;
typedef __attribute__((ext_vector_type(4)))  int    v4i;

#if defined(__has_builtin)
#if __has_builtin(__builtin_amdgcn_global_load_async_to_lds_b128) && \
    __has_builtin(__builtin_amdgcn_s_wait_asynccnt)
#define USE_ASYNC_LDS 1
#endif
#endif
#ifndef USE_ASYNC_LDS
#define USE_ASYNC_LDS 0
#endif

#if USE_ASYNC_LDS
// Builtin parameter types (from clang diagnostic): int4 in address_space(1)
// (printed as __device__) for the global source; LDS side is address_space(3).
typedef __attribute__((address_space(1))) v4i global_v4i;
typedef __attribute__((address_space(3))) v4i shared_v4i;

// Direct global->LDS async copy (no VGPR round trip). The uintptr casts are
// folded by InstCombine back to the original addrspace(1)/(3) pointers.
static __device__ __forceinline__ void async_b128(void* lds_dst,
                                                  const void* gsrc) {
  __builtin_amdgcn_global_load_async_to_lds_b128(
      (global_v4i*)(unsigned long long)gsrc,
      (shared_v4i*)(unsigned long long)lds_dst, 0, 0);
}
#endif

// ---- WMMA helpers -----------------------------------------------------------

// A-matrix 16x32 bf16 fragment from LDS (row-major, pitch in elements).
// Lanes 0-15 = rows m, K = {0..15}; lanes 16-31 = rows m,
// K = {16..31} (matches packed B). Two 16B LDS loads per lane.
static __device__ __forceinline__ v16bf lds_a_frag(const __bf16* p, int pitch,
                                                   int k0, int lane) {
  const int m  = lane & 15;
  const int kh = lane >> 4;
  const __bf16* p0 = p + m * pitch + k0 + kh * 16;
  v8bf lo = *(const v8bf*)p0;
  v8bf hi = *(const v8bf*)(p0 + 8);
  return __builtin_shufflevector(lo, hi, 0, 1, 2, 3, 4, 5, 6, 7,
                                 8, 9, 10, 11, 12, 13, 14, 15);
}

// B-matrix 32x16 bf16 fragment from a pre-packed weight image: fragment (kt,nt)
// for `lane` is 16 contiguous bf16 (32 bytes) -> single wide global load, L2-hot.
static __device__ __forceinline__ v16bf gbl_b_frag(const __bf16* pk, int KT,
                                                   int kt, int nt, int lane) {
  return *(const v16bf*)(pk + ((((size_t)nt * KT) + kt) * 32 + lane) * 16);
}

static __device__ __forceinline__ v8f wmma_bf16(v16bf a, v16bf b, v8f c) {
  return __builtin_amdgcn_wmma_f32_16x16x32_bf16(false, a, false, b,
                                                 (short)0, c, false, false);
}

// ---- one-time kernels -------------------------------------------------------

// Pack row-major f32 W[K][N] into per-lane-contiguous bf16 B fragments.
// lane = (n%16) + 16*((k%32)/16); element kk = k%16 (pairs per VGPR).
__global__ __launch_bounds__(256) void k_pack_b(const float* __restrict__ W,
                                                __bf16* __restrict__ out,
                                                int K, int N) {
  int idx = blockIdx.x * 256 + threadIdx.x;
  if (idx >= K * N) return;
  int k = idx / N, n = idx - k * N;
  int kt = k >> 5, nt = n >> 4;
  int lane = (n & 15) + (((k >> 4) & 1) << 4);
  int kk = k & 15;
  size_t off = ((((size_t)nt * (K >> 5)) + kt) * 32 + lane) * 16 + kk;
  out[off] = (__bf16)W[(size_t)k * N + n];
}

__global__ __launch_bounds__(256) void k_init(float* __restrict__ prev,
                                              float* __restrict__ acc) {
  int i = blockIdx.x * 256 + threadIdx.x;
  if (i < B_ * D_) prev[i] = 0.f;
  if (i < B_ * OUT_) acc[i] = 0.f;
}

// ---- per-step kernel 1: x=x_t+prev; v=x@wv+bv; a=LN(v@wo+bo+x) --------------

__global__ __launch_bounds__(256) void k_attn(
    const float* __restrict__ lhs, int t, const float* __restrict__ prev,
    const __bf16* __restrict__ wv_pk, const float* __restrict__ bv,
    const __bf16* __restrict__ wo_pk, const float* __restrict__ bo,
    const float* __restrict__ ln1_g, const float* __restrict__ ln1_b,
    float* __restrict__ a_f32, __bf16* __restrict__ a_bf16) {
  __shared__ __align__(16) __bf16 xb[16 * 784];  // x in bf16, padded pitch
  __shared__ __align__(16) __bf16 vb[16 * 784];  // v in bf16
  __shared__ float rowsum[16], rowsq[16];

  const int tid = threadIdx.x;
  const int lane = tid & 31;
  const int w = tid >> 5;
  const int row0 = blockIdx.x * 16;

  if (tid < 16) { rowsum[tid] = 0.f; rowsq[tid] = 0.f; }

  // Phase 1: x = x_t + h_{t-1} -> LDS bf16 (float4 streaming loads).
  for (int i = tid; i < 16 * 192; i += 256) {
    int m = i / 192, c4 = i - m * 192;
    float4 xv = *(const float4*)&lhs[(((size_t)(row0 + m)) * S_ + t) * D_ + c4 * 4];
    float4 pv = *(const float4*)&prev[(size_t)(row0 + m) * D_ + c4 * 4];
    __bf16* dst = &xb[m * 784 + c4 * 4];
    dst[0] = (__bf16)(xv.x + pv.x);
    dst[1] = (__bf16)(xv.y + pv.y);
    dst[2] = (__bf16)(xv.z + pv.z);
    dst[3] = (__bf16)(xv.w + pv.w);
  }
  __syncthreads();

  // Phase 2: v = x @ wv + bv (48 N-tiles, 6 per wave, K = 24 WMMAs).
  for (int it = 0; it < 6; ++it) {
    int nt = w * 6 + it;
    v8f c = {};
    for (int kt = 0; kt < 24; ++kt) {
      v16bf a = lds_a_frag(xb, 784, kt * 32, lane);
      v16bf b = gbl_b_frag(wv_pk, 24, kt, nt, lane);
      c = wmma_bf16(a, b, c);
    }
    int n = nt * 16 + (lane & 15);
    float bvn = bv[n];
    int mb = (lane >> 4) * 8;  // C/D layout: VGPR r -> m = r + 8*(lane>=16)
    for (int r = 0; r < 8; ++r)
      vb[(mb + r) * 784 + n] = (__bf16)(c[r] + bvn);
  }
  __syncthreads();

  // Phase 3: pre-LN = v @ wo + bo + x ; per-row stats via ds_add_f32.
  v8f cacc[6] = {};
  for (int it = 0; it < 6; ++it) {
    int nt = w * 6 + it;
    v8f c = {};
    for (int kt = 0; kt < 24; ++kt) {
      v16bf a = lds_a_frag(vb, 784, kt * 32, lane);
      v16bf b = gbl_b_frag(wo_pk, 24, kt, nt, lane);
      c = wmma_bf16(a, b, c);
    }
    int n = nt * 16 + (lane & 15);
    float bon = bo[n];
    int mb = (lane >> 4) * 8;
    for (int r = 0; r < 8; ++r) {
      int m = mb + r;
      float xres = lhs[(((size_t)(row0 + m)) * S_ + t) * D_ + n] +
                   prev[(size_t)(row0 + m) * D_ + n];
      float v = c[r] + bon + xres;
      c[r] = v;
      atomicAdd(&rowsum[m], v);
      atomicAdd(&rowsq[m], v * v);
    }
    cacc[it] = c;
  }
  __syncthreads();

  // Phase 4: LayerNorm -> a (f32 for residual, bf16 for next GEMM's A).
  for (int it = 0; it < 6; ++it) {
    int nt = w * 6 + it;
    int n = nt * 16 + (lane & 15);
    float g = ln1_g[n], bb = ln1_b[n];
    int mb = (lane >> 4) * 8;
    for (int r = 0; r < 8; ++r) {
      int m = mb + r;
      float mean = rowsum[m] * (1.f / 768.f);
      float var = rowsq[m] * (1.f / 768.f) - mean * mean;
      float av = (cacc[it][r] - mean) * rsqrtf(var + EPS_) * g + bb;
      a_f32[(size_t)(row0 + m) * D_ + n] = av;
      a_bf16[(size_t)(row0 + m) * D_ + n] = (__bf16)av;
    }
  }
}

// ---- per-step kernel 2: inter = gelu(a @ wi + bi) ---------------------------

__global__ __launch_bounds__(256) void k_ffn1(
    const __bf16* __restrict__ a_bf16, const __bf16* __restrict__ wi_pk,
    const float* __restrict__ bi, __bf16* __restrict__ inter) {
  __shared__ __align__(16) __bf16 ab[16 * 784];
  const int tid = threadIdx.x, lane = tid & 31, w = tid >> 5;
  const int row0 = blockIdx.x * 16;
  const int ntile0 = blockIdx.y * 16;  // 256 output cols per block

#if USE_ASYNC_LDS
  for (int i = tid; i < 16 * 96; i += 256) {  // direct global->LDS, no VGPRs
    int m = i / 96, c8 = i - m * 96;
    async_b128(&ab[m * 784 + c8 * 8],
               &a_bf16[(size_t)(row0 + m) * D_ + c8 * 8]);
  }
  __builtin_amdgcn_s_wait_asynccnt(0);
#else
  for (int i = tid; i < 16 * 96; i += 256) {  // 16B chunks
    int m = i / 96, c8 = i - m * 96;
    *(uint4*)&ab[m * 784 + c8 * 8] =
        *(const uint4*)&a_bf16[(size_t)(row0 + m) * D_ + c8 * 8];
  }
#endif
  __syncthreads();

  for (int it = 0; it < 2; ++it) {
    int nt = ntile0 + w * 2 + it;
    v8f c = {};
    for (int kt = 0; kt < 24; ++kt) {
      v16bf a = lds_a_frag(ab, 784, kt * 32, lane);
      v16bf b = gbl_b_frag(wi_pk, 24, kt, nt, lane);
      c = wmma_bf16(a, b, c);
    }
    int n = nt * 16 + (lane & 15);
    float bin = bi[n];
    int mb = (lane >> 4) * 8;
    for (int r = 0; r < 8; ++r) {
      float x = c[r] + bin;
      float ge = 0.5f * x * (1.f + erff(x * 0.70710678f));  // exact gelu
      inter[(size_t)(row0 + mb + r) * FF_ + n] = (__bf16)ge;
    }
  }
}

// ---- per-step kernel 3: h = LN(inter@w2+b2+a); acc += softmax(h@fc_w+fc_b) --

__global__ __launch_bounds__(256) void k_ffn2(
    const __bf16* __restrict__ inter, const __bf16* __restrict__ w2_pk,
    const float* __restrict__ b2, const float* __restrict__ a_f32,
    const float* __restrict__ ln2_g, const float* __restrict__ ln2_b,
    const float* __restrict__ fc_w, const float* __restrict__ fc_b,
    float* __restrict__ h_out, float* __restrict__ acc) {
  // 48KB aliased: K-chunk of inter (bf16 [16][1040]) then h (f32 [16][768]).
  __shared__ __align__(16) char smem[16 * 768 * 4];
  __shared__ float rowsum[16], rowsq[16], rowsum2[16];
  __shared__ float logits[16 * OUT_];
  __bf16* ib = (__bf16*)smem;
  float* h_sh = (float*)smem;
  const int tid = threadIdx.x, lane = tid & 31, w = tid >> 5;
  const int row0 = blockIdx.x * 16;

  if (tid < 16) { rowsum[tid] = 0.f; rowsq[tid] = 0.f; rowsum2[tid] = 0.f; }

  v8f cacc[6] = {};
  for (int kc = 0; kc < 3; ++kc) {  // K = 3072 in chunks of 1024
    __syncthreads();
#if USE_ASYNC_LDS
    for (int i = tid; i < 16 * 128; i += 256) {
      int m = i / 128, c8 = i - m * 128;
      async_b128(&ib[m * 1040 + c8 * 8],
                 &inter[(size_t)(row0 + m) * FF_ + kc * 1024 + c8 * 8]);
    }
    __builtin_amdgcn_s_wait_asynccnt(0);
#else
    for (int i = tid; i < 16 * 128; i += 256) {
      int m = i / 128, c8 = i - m * 128;
      *(uint4*)&ib[m * 1040 + c8 * 8] =
          *(const uint4*)&inter[(size_t)(row0 + m) * FF_ + kc * 1024 + c8 * 8];
    }
#endif
    __syncthreads();
    for (int it = 0; it < 6; ++it) {
      int nt = w * 6 + it;
      v8f c = cacc[it];
      for (int ktl = 0; ktl < 32; ++ktl) {
        v16bf a = lds_a_frag(ib, 1040, ktl * 32, lane);
        v16bf b = gbl_b_frag(w2_pk, 96, kc * 32 + ktl, nt, lane);
        c = wmma_bf16(a, b, c);
      }
      cacc[it] = c;
    }
  }
  __syncthreads();

  // bias + residual + LN stats
  for (int it = 0; it < 6; ++it) {
    int nt = w * 6 + it, n = nt * 16 + (lane & 15);
    float b2n = b2[n];
    int mb = (lane >> 4) * 8;
    for (int r = 0; r < 8; ++r) {
      int m = mb + r;
      float v = cacc[it][r] + b2n + a_f32[(size_t)(row0 + m) * D_ + n];
      cacc[it][r] = v;
      atomicAdd(&rowsum[m], v);
      atomicAdd(&rowsq[m], v * v);
    }
  }
  __syncthreads();

  // LayerNorm -> h (to LDS for the FC dot, to global as next step's carry)
  for (int it = 0; it < 6; ++it) {
    int nt = w * 6 + it, n = nt * 16 + (lane & 15);
    float g = ln2_g[n], bb = ln2_b[n];
    int mb = (lane >> 4) * 8;
    for (int r = 0; r < 8; ++r) {
      int m = mb + r;
      float mean = rowsum[m] * (1.f / 768.f);
      float var = rowsq[m] * (1.f / 768.f) - mean * mean;
      float h = (cacc[it][r] - mean) * rsqrtf(var + EPS_) * g + bb;
      h_sh[m * D_ + n] = h;
      h_out[(size_t)(row0 + m) * D_ + n] = h;
      atomicAdd(&rowsum2[m], h);
    }
  }
  __syncthreads();

  // logits = h @ fc_w + fc_b  (160 threads: one per (m, o))
  if (tid < 16 * OUT_) {
    int m = tid / OUT_, o = tid - m * OUT_;
    float s = fc_b[o];
    for (int n = 0; n < D_; ++n) s += h_sh[m * D_ + n] * fc_w[n * OUT_ + o];
    logits[tid] = s;
  }
  __syncthreads();

  // masked softmax accumulate (row owned exclusively by this WG this step)
  if (tid < 16) {
    int m = tid;
    if (rowsum2[m] != 0.f) {
      float mx = logits[m * OUT_];
      for (int o = 1; o < OUT_; ++o) mx = fmaxf(mx, logits[m * OUT_ + o]);
      float e[OUT_], se = 0.f;
      for (int o = 0; o < OUT_; ++o) {
        e[o] = __expf(logits[m * OUT_ + o] - mx);
        se += e[o];
      }
      float inv = 1.f / se;
      for (int o = 0; o < OUT_; ++o)
        acc[(size_t)(row0 + m) * OUT_ + o] += e[o] * inv;
    }
  }
}

// ---- final: normalize + pad-row mask ---------------------------------------

__global__ __launch_bounds__(128) void k_final(
    const long long* __restrict__ ids, const long long* __restrict__ msk,
    const float* __restrict__ acc, float* __restrict__ out) {
  int b = threadIdx.x;
  bool any_pad = false;  // exists position with id==0 && mask==0
  for (int s = 0; s < S_; ++s)
    if (ids[(size_t)b * S_ + s] == 0 && msk[(size_t)b * S_ + s] == 0) {
      any_pad = true;
      break;
    }
  float ssum = 0.f;
  for (int o = 0; o < OUT_; ++o) ssum += acc[b * OUT_ + o];
  float inv = (ssum != 0.f) ? 1.f / ssum : 0.f;
  for (int o = 0; o < OUT_; ++o)
    out[b * OUT_ + o] = any_pad ? acc[b * OUT_ + o] * inv : 0.f;
}

// ---- host -------------------------------------------------------------------

extern "C" void kernel_launch(void* const* d_in, const int* in_sizes, int n_in,
                              void* d_out, int out_size, void* d_ws,
                              size_t ws_size, hipStream_t stream) {
  (void)in_sizes; (void)n_in; (void)out_size; (void)ws_size;
  const float* lhs = (const float*)d_in[0];
  const long long* ids = (const long long*)d_in[1];  // int64 in reference
  const long long* msk = (const long long*)d_in[2];
  // d_in[3..6] (wq,bq,wk,bk) are provably dead: softmax over 1 key == 1.0.
  const float* wv = (const float*)d_in[7];
  const float* bv = (const float*)d_in[8];
  const float* wo = (const float*)d_in[9];
  const float* bo = (const float*)d_in[10];
  const float* ln1g = (const float*)d_in[11];
  const float* ln1b = (const float*)d_in[12];
  const float* wi = (const float*)d_in[13];
  const float* bi = (const float*)d_in[14];
  const float* w2 = (const float*)d_in[15];
  const float* b2 = (const float*)d_in[16];
  const float* ln2g = (const float*)d_in[17];
  const float* ln2b = (const float*)d_in[18];
  const float* fcw = (const float*)d_in[19];
  const float* fcb = (const float*)d_in[20];

  char* ws = (char*)d_ws;
  size_t off = 0;
  auto alloc = [&](size_t bytes) -> void* {
    void* p = ws + off;
    off = (off + bytes + 255) & ~(size_t)255;
    return p;
  };
  __bf16* wv_pk = (__bf16*)alloc((size_t)D_ * D_ * 2);
  __bf16* wo_pk = (__bf16*)alloc((size_t)D_ * D_ * 2);
  __bf16* wi_pk = (__bf16*)alloc((size_t)D_ * FF_ * 2);
  __bf16* w2_pk = (__bf16*)alloc((size_t)FF_ * D_ * 2);
  float* prev = (float*)alloc((size_t)B_ * D_ * 4);
  float* a32 = (float*)alloc((size_t)B_ * D_ * 4);
  __bf16* a16 = (__bf16*)alloc((size_t)B_ * D_ * 2);
  __bf16* inter16 = (__bf16*)alloc((size_t)B_ * FF_ * 2);
  float* acc = (float*)alloc((size_t)B_ * OUT_ * 4);

  // One-time (per launch, deterministic): pack weights to bf16 B-layout, zero state.
  k_pack_b<<<(D_ * D_ + 255) / 256, 256, 0, stream>>>(wv, wv_pk, D_, D_);
  k_pack_b<<<(D_ * D_ + 255) / 256, 256, 0, stream>>>(wo, wo_pk, D_, D_);
  k_pack_b<<<(D_ * FF_ + 255) / 256, 256, 0, stream>>>(wi, wi_pk, D_, FF_);
  k_pack_b<<<(FF_ * D_ + 255) / 256, 256, 0, stream>>>(w2, w2_pk, FF_, D_);
  k_init<<<(B_ * D_ + 255) / 256, 256, 0, stream>>>(prev, acc);

  for (int t = 0; t < S_; ++t) {
    k_attn<<<8, 256, 0, stream>>>(lhs, t, prev, wv_pk, bv, wo_pk, bo, ln1g,
                                  ln1b, a32, a16);
    k_ffn1<<<dim3(8, 12), 256, 0, stream>>>(a16, wi_pk, bi, inter16);
    k_ffn2<<<8, 256, 0, stream>>>(inter16, w2_pk, b2, a32, ln2g, ln2b, fcw,
                                  fcb, prev, acc);
  }
  k_final<<<1, 128, 0, stream>>>(ids, msk, acc, (float*)d_out);
}
